// VariableRVQ_18468359373176
// MI455X (gfx1250) — compile-verified, matrix operand order checked by reading, not run on hardware
//
#include <hip/hip_runtime.h>
#include <math.h>

// ---------------------------------------------------------------------------
// Residual VQ (4 layers) for MI455X / gfx1250 (wave32, WMMA bf16 16x16x32).
//
// d_out layout (floats, in reference return order):
//   [0,          N*D)          quantized_out   (used as residual scratch first)
//   [N*D,        N*D+4N)       indices [N,4] as float
//   [+0,  +4)  loss per layer   [+4, +8) perplexity   [+8, +12) mse
//
// d_ws layout:
//   ushort cb_bf16 [totRows*128] ; float e2[totRows] ; float counts[totRows] ;
//   float sumsq[4]
// ---------------------------------------------------------------------------

typedef __attribute__((ext_vector_type(16))) __bf16 v16bf;
typedef __attribute__((ext_vector_type(8)))  float  v8f;

union ABFrag {
    v16bf          v;
    unsigned short u[16];
    uint4          q[2];
};

__device__ __forceinline__ unsigned short f2bf(float f) {
    unsigned int u = __float_as_uint(f);
    unsigned int r = u + 0x7FFFu + ((u >> 16) & 1u);   // round-to-nearest-even
    return (unsigned short)(r >> 16);
}

// ---------------------------------------------------------------------------
__global__ void vq_init_kernel(float* __restrict__ p, int n) {
    int i = blockIdx.x * blockDim.x + threadIdx.x;
    int stride = gridDim.x * blockDim.x;
    for (; i < n; i += stride) p[i] = 0.0f;
}

// ---------------------------------------------------------------------------
// Convert one codebook to bf16 + precompute ||e||^2.  One wave per code row.
__global__ void vq_prep_kernel(const float* __restrict__ cb,
                               unsigned short* __restrict__ dst,
                               float* __restrict__ e2, int K) {
    int tid  = threadIdx.x;
    int lane = tid & 31;
    int row  = blockIdx.x * 8 + (tid >> 5);
    if (row >= K) return;

    float4 v = ((const float4*)(cb + (size_t)row * 128))[lane];
    unsigned short b0 = f2bf(v.x), b1 = f2bf(v.y), b2 = f2bf(v.z), b3 = f2bf(v.w);
    uint2 packed;
    packed.x = (unsigned)b0 | ((unsigned)b1 << 16);
    packed.y = (unsigned)b2 | ((unsigned)b3 << 16);
    ((uint2*)(dst + (size_t)row * 128))[lane] = packed;

    float ss = v.x * v.x + v.y * v.y + v.z * v.z + v.w * v.w;
    #pragma unroll
    for (int m = 16; m >= 1; m >>= 1) ss += __shfl_xor(ss, m, 32);
    if (lane == 0) e2[row] = ss;
}

// ---------------------------------------------------------------------------
// Load a 16x128 bf16 A tile (4 chunks of 16x32) in CDNA5 A-fragment layout.
__device__ __forceinline__ void load_a_tile(ABFrag a[4], const float* rrow, int kb) {
    #pragma unroll
    for (int c = 0; c < 4; ++c) {
        float4 f0 = *(const float4*)(rrow + c * 32 + kb);
        float4 f1 = *(const float4*)(rrow + c * 32 + kb + 4);
        float4 f2 = *(const float4*)(rrow + c * 32 + 16 + kb);
        float4 f3 = *(const float4*)(rrow + c * 32 + 16 + kb + 4);
        a[c].u[0]  = f2bf(f0.x); a[c].u[1]  = f2bf(f0.y);
        a[c].u[2]  = f2bf(f0.z); a[c].u[3]  = f2bf(f0.w);
        a[c].u[4]  = f2bf(f1.x); a[c].u[5]  = f2bf(f1.y);
        a[c].u[6]  = f2bf(f1.z); a[c].u[7]  = f2bf(f1.w);
        a[c].u[8]  = f2bf(f2.x); a[c].u[9]  = f2bf(f2.y);
        a[c].u[10] = f2bf(f2.z); a[c].u[11] = f2bf(f2.w);
        a[c].u[12] = f2bf(f3.x); a[c].u[13] = f2bf(f3.y);
        a[c].u[14] = f2bf(f3.z); a[c].u[15] = f2bf(f3.w);
    }
}

// ---------------------------------------------------------------------------
// One RVQ layer.  Block: 256 threads = 8 waves.  Each wave owns TWO 16-row
// A tiles (32 rows) so every LDS-resident B fragment feeds two independent
// WMMA chains (halves ds traffic per wmma, hides D->C accumulation latency).
// Block owns 256 rows.
__global__ void __launch_bounds__(256)
vq_layer_kernel(const float* __restrict__ resid_in,
                float* __restrict__ resid_out,
                const float* __restrict__ cbf32,
                const unsigned short* __restrict__ cb16,
                const float* __restrict__ e2,
                float* __restrict__ counts,
                float* __restrict__ sumsq,      // single accumulator for layer
                float* __restrict__ idx_out,    // [N,4] as float
                int K, int layer) {
    __shared__ __align__(16) unsigned short lcb[128 * 128]; // 32 KB bf16 code tile
    __shared__ float le2[128];

    const int tid  = threadIdx.x;
    const int lane = tid & 31;
    const int wave = tid >> 5;
    const int l15  = lane & 15;
    const int kb   = (lane >> 4) * 8;   // A/B fragment K-base per half-wave

    const int rowBase = blockIdx.x * 256 + wave * 32;
    const int row0    = rowBase + l15;        // A tile 0 row (M = l15)
    const int row1    = rowBase + 16 + l15;   // A tile 1 row

    ABFrag a0[4], a1[4];
    load_a_tile(a0, resid_in + (size_t)row0 * 128, kb);
    load_a_tile(a1, resid_in + (size_t)row1 * 128, kb);

    float best0[8], best1[8];
    int   bidx0[8], bidx1[8];
    #pragma unroll
    for (int j = 0; j < 8; ++j) {
        best0[j] = 3.0e38f; bidx0[j] = 0;
        best1[j] = 3.0e38f; bidx1[j] = 0;
    }

    // ---- Sweep codebook in 128-code LDS stages ----
    const int stages = K >> 7;
    for (int s = 0; s < stages; ++s) {
        const int codeBase = s << 7;
        __syncthreads();
        // cooperative stage: 128 codes x 128 bf16 = 2048 uint4
        const uint4* src  = (const uint4*)(cb16 + (size_t)codeBase * 128);
        uint4*       dstl = (uint4*)lcb;
        #pragma unroll
        for (int i = 0; i < 8; ++i) dstl[tid + (i << 8)] = src[tid + (i << 8)];
        if (tid < 128) le2[tid] = e2[codeBase + tid];
        __syncthreads();

        #pragma unroll 2
        for (int t = 0; t < 8; ++t) {               // 8 column tiles of 16 codes
            const int cl = (t << 4) + l15;           // local code = B column
            const unsigned short* brow = lcb + (size_t)cl * 128;
            v8f acc0 = {0.f, 0.f, 0.f, 0.f, 0.f, 0.f, 0.f, 0.f};
            v8f acc1 = {0.f, 0.f, 0.f, 0.f, 0.f, 0.f, 0.f, 0.f};
            #pragma unroll
            for (int c = 0; c < 4; ++c) {            // D=128 -> 4 x K=32
                ABFrag b;
                b.q[0] = *(const uint4*)(brow + c * 32 + kb);
                b.q[1] = *(const uint4*)(brow + c * 32 + 16 + kb);
                acc0 = __builtin_amdgcn_wmma_f32_16x16x32_bf16(
                        false, a0[c].v, false, b.v, (short)0, acc0, false, false);
                acc1 = __builtin_amdgcn_wmma_f32_16x16x32_bf16(
                        false, a1[c].v, false, b.v, (short)0, acc1, false, false);
            }
            const int   code = codeBase + cl;
            const float ev   = le2[cl];
            #pragma unroll
            for (int j = 0; j < 8; ++j) {
                float sc0 = ev - 2.0f * acc0[j];     // ||e||^2 - 2 r.e
                bool  c0  = sc0 < best0[j];
                best0[j] = c0 ? sc0 : best0[j];
                bidx0[j] = c0 ? code : bidx0[j];
                float sc1 = ev - 2.0f * acc1[j];
                bool  c1  = sc1 < best1[j];
                best1[j] = c1 ? sc1 : best1[j];
                bidx1[j] = c1 ? code : bidx1[j];
            }
        }
    }

    // ---- Cross-lane argmin (masks <16 keep half-wave row groups separate) ----
    #pragma unroll
    for (int j = 0; j < 8; ++j) {
        #pragma unroll
        for (int m = 8; m >= 1; m >>= 1) {
            float ov0 = __shfl_xor(best0[j], m, 32);
            int   oi0 = __shfl_xor(bidx0[j], m, 32);
            bool  t0  = (ov0 < best0[j]) || (ov0 == best0[j] && oi0 < bidx0[j]);
            best0[j] = t0 ? ov0 : best0[j];
            bidx0[j] = t0 ? oi0 : bidx0[j];
            float ov1 = __shfl_xor(best1[j], m, 32);
            int   oi1 = __shfl_xor(bidx1[j], m, 32);
            bool  t1  = (ov1 < best1[j]) || (ov1 == best1[j] && oi1 < bidx1[j]);
            best1[j] = t1 ? ov1 : best1[j];
            bidx1[j] = t1 ? oi1 : bidx1[j];
        }
    }

    // ---- Emit indices + usage counts ----
    // tile0: lane0 -> rows 0..7, lane16 -> rows 8..15
    // tile1: lane0 -> rows 16..23, lane16 -> rows 24..31
    if (lane == 0) {
        #pragma unroll
        for (int j = 0; j < 8; ++j) {
            int gr0 = rowBase + j;
            idx_out[(size_t)gr0 * 4 + layer] = (float)bidx0[j];
            atomicAdd(&counts[bidx0[j]], 1.0f);
            int gr1 = rowBase + 16 + j;
            idx_out[(size_t)gr1 * 4 + layer] = (float)bidx1[j];
            atomicAdd(&counts[bidx1[j]], 1.0f);
        }
    } else if (lane == 16) {
        #pragma unroll
        for (int j = 0; j < 8; ++j) {
            int gr0 = rowBase + 8 + j;
            idx_out[(size_t)gr0 * 4 + layer] = (float)bidx0[j];
            atomicAdd(&counts[bidx0[j]], 1.0f);
            int gr1 = rowBase + 24 + j;
            idx_out[(size_t)gr1 * 4 + layer] = (float)bidx1[j];
            atomicAdd(&counts[bidx1[j]], 1.0f);
        }
    }

    // ---- fp32 residual update + sum of squares (loss & mse share it) ----
    float local_ss = 0.0f;
    #pragma unroll
    for (int r = 0; r < 32; ++r) {
        const int j    = r & 7;
        const int srcl = (r & 8) ? 16 : 0;
        const int ridx = (r < 16) ? __shfl(bidx0[j], srcl, 32)
                                  : __shfl(bidx1[j], srcl, 32);
        const int grow = rowBase + r;
        float4 cv = *(const float4*)(cbf32    + (size_t)ridx * 128 + lane * 4);
        float4 rv = *(const float4*)(resid_in + (size_t)grow * 128 + lane * 4);
        float4 nr;
        nr.x = rv.x - cv.x; nr.y = rv.y - cv.y;
        nr.z = rv.z - cv.z; nr.w = rv.w - cv.w;
        *(float4*)(resid_out + (size_t)grow * 128 + lane * 4) = nr;
        local_ss += nr.x * nr.x + nr.y * nr.y + nr.z * nr.z + nr.w * nr.w;
    }
    #pragma unroll
    for (int m = 16; m >= 1; m >>= 1) local_ss += __shfl_xor(local_ss, m, 32);
    if (lane == 0) atomicAdd(sumsq, local_ss);
}

// ---------------------------------------------------------------------------
// quantized_out = x - residual_final   (in place over the residual region)
__global__ void vq_finalize_kernel(const float* __restrict__ x,
                                   float* __restrict__ out, int n4) {
    int i = blockIdx.x * blockDim.x + threadIdx.x;
    int stride = gridDim.x * blockDim.x;
    const float4* x4 = (const float4*)x;
    float4*       o4 = (float4*)out;
    for (; i < n4; i += stride) {
        float4 xv = x4[i];
        float4 rv = o4[i];
        float4 q;
        q.x = xv.x - rv.x; q.y = xv.y - rv.y;
        q.z = xv.z - rv.z; q.w = xv.w - rv.w;
        o4[i] = q;
    }
}

// ---------------------------------------------------------------------------
// Per-layer scalars: loss = 1.25*ss/(N*D); mse = ss/(N*D); perplexity.
__global__ void vq_stats_kernel(const float* __restrict__ counts,
                                const float* __restrict__ sumsq,
                                float* __restrict__ loss_out,
                                float* __restrict__ perp_out,
                                float* __restrict__ mse_out,
                                int K, float invN, float invND) {
    __shared__ float sdata[256];
    int tid = threadIdx.x;
    float acc = 0.0f;
    for (int b = tid; b < K; b += 256) {
        float p = counts[b] * invN;
        acc += p * logf(p + 1e-10f);
    }
    sdata[tid] = acc;
    __syncthreads();
    for (int off = 128; off > 0; off >>= 1) {
        if (tid < off) sdata[tid] += sdata[tid + off];
        __syncthreads();
    }
    if (tid == 0) {
        float ss = sumsq[0];
        loss_out[0] = 1.25f * ss * invND;
        mse_out[0]  = ss * invND;
        perp_out[0] = expf(-sdata[0]);
    }
}

// ---------------------------------------------------------------------------
extern "C" void kernel_launch(void* const* d_in, const int* in_sizes, int n_in,
                              void* d_out, int out_size, void* d_ws, size_t ws_size,
                              hipStream_t stream) {
    const int D = 128;
    const float* x = (const float*)d_in[0];
    const float* cbs[4] = {(const float*)d_in[1], (const float*)d_in[2],
                           (const float*)d_in[3], (const float*)d_in[4]};
    const int N = in_sizes[0] / D;            // 131072
    int K[4], rowoff[4], tot = 0;
    for (int i = 0; i < 4; ++i) { K[i] = in_sizes[i + 1] / D; rowoff[i] = tot; tot += K[i]; }

    unsigned short* ws_cb   = (unsigned short*)d_ws;
    float* ws_e2     = (float*)((char*)d_ws + (size_t)tot * D * 2);
    float* ws_counts = ws_e2 + tot;
    float* ws_sumsq  = ws_counts + tot;

    float* outp = (float*)d_out;
    const size_t ND = (size_t)N * D;
    float* idx_out  = outp + ND;
    float* loss_out = outp + ND + (size_t)N * 4;
    float* perp_out = loss_out + 4;
    float* mse_out  = perp_out + 4;

    // zero counts + sumsq (every call)
    vq_init_kernel<<<8, 256, 0, stream>>>(ws_counts, tot + 4);

    // bf16 codebooks + ||e||^2
    for (int i = 0; i < 4; ++i)
        vq_prep_kernel<<<K[i] / 8, 256, 0, stream>>>(
            cbs[i], ws_cb + (size_t)rowoff[i] * D, ws_e2 + rowoff[i], K[i]);

    // 4 sequential RVQ layers; residual lives in d_out[0..N*D)
    const float* rin = x;
    for (int l = 0; l < 4; ++l) {
        vq_layer_kernel<<<N / 256, 256, 0, stream>>>(
            rin, outp, cbs[l],
            ws_cb + (size_t)rowoff[l] * D, ws_e2 + rowoff[l],
            ws_counts + rowoff[l], ws_sumsq + l,
            idx_out, K[l], l);
        rin = outp;
    }

    // quantized_out = x - residual
    vq_finalize_kernel<<<4096, 256, 0, stream>>>(x, outp, (int)(ND / 4));

    // per-layer scalars
    const float invN  = 1.0f / (float)N;
    const float invND = 1.0f / (float)ND;
    for (int l = 0; l < 4; ++l)
        vq_stats_kernel<<<1, 256, 0, stream>>>(
            ws_counts + rowoff[l], ws_sumsq + l,
            loss_out + l, perp_out + l, mse_out + l, K[l], invN, invND);
}
